// LSTM_58360015618742
// MI455X (gfx1250) — matrix operand order repaired and checked by reading, not compiled, not run on hardware
//
#include <hip/hip_runtime.h>
#include <hip/hip_bf16.h>
#include <stdint.h>

// ---------------------------------------------------------------------------
// LSTM + LayerNorm + skip for MI455X (gfx1250, wave32, WMMA).
//   B=512, T=256, I=H=256, G=4H=1024, combined K = I+H = 512.
// Kernel 1: pack W_ih|W_hh -> bf16 Wc[1024][512] (row-major, N x K), bias sum.
// Kernel 2: persistent scan. 32 WGs x 256 threads; each WG owns 16 batch rows.
//           Per step: [16x1024] gates via v_wmma_f32_16x16x32_bf16, K=512,
//           then fused LSTM cell + LayerNorm + skip-add, state kept in LDS.
// Kernel 3: broadcast hT/cT across T (pure bandwidth).
// Workspace use: 1MB Wc + 4KB bias + 2x512KB hT/cT  (~2.1 MB).
// ---------------------------------------------------------------------------

typedef __attribute__((ext_vector_type(16))) __bf16 v16bf;
typedef __attribute__((ext_vector_type(8)))  float  v8f;

#define BN 512
#define TT 256
#define II 256
#define HH 256
#define GG 1024   // 4*H
#define KK 512    // I + H

// LDS strides (elements); rows 16B aligned, padded vs 64-bank conflicts.
#define GSTRIDE 1028   // floats per gate row
#define ASTRIDE 520    // bf16 per [x|h] row
#define XSTRIDE 260    // floats per x/c/h row

// LDS layout (bytes)
#define OFF_GATES 0
#define SZ_GATES  (16 * GSTRIDE * 4)            // 65792
#define OFF_AXH   (OFF_GATES + SZ_GATES)
#define SZ_AXH    (16 * ASTRIDE * 2)            // 16640
#define OFF_X     (OFF_AXH + SZ_AXH)
#define SZ_ROW32  (16 * XSTRIDE * 4)            // 16640
#define OFF_C     (OFF_X + SZ_ROW32)
#define OFF_H     (OFF_C + SZ_ROW32)
#define OFF_BIAS  (OFF_H + SZ_ROW32)
#define SZ_BIAS   (GG * 4)
#define LDS_BYTES (OFF_BIAS + SZ_BIAS)          // 136448

__device__ __forceinline__ float fsig(float v)  { return 1.0f / (1.0f + __expf(-v)); }
__device__ __forceinline__ float ftanh(float v) { return 1.0f - 2.0f / (1.0f + __expf(2.0f * v)); }

// ---------------------------------------------------------------------------
// Kernel 1: weight packing + bias sum
// ---------------------------------------------------------------------------
__global__ void lstm_prep_kernel(const float* __restrict__ Wih,
                                 const float* __restrict__ Whh,
                                 const float* __restrict__ bih,
                                 const float* __restrict__ bhh,
                                 __bf16* __restrict__ Wc,
                                 float* __restrict__ biasSum) {
  const int gid = blockIdx.x * 256 + threadIdx.x;     // 0 .. GG*KK-1
  const int n = gid >> 9;                              // row (gate output col)
  const int k = gid & 511;
  const float v = (k < II) ? Wih[n * II + k] : Whh[n * HH + (k - II)];
  Wc[gid] = (__bf16)v;
  if (gid < GG) biasSum[gid] = bih[gid] + bhh[gid];
}

// ---------------------------------------------------------------------------
// Kernel 2: persistent LSTM scan with fused LN + skip
// ---------------------------------------------------------------------------
__global__ __launch_bounds__(256)
void lstm_scan_kernel(const float* __restrict__ x,
                      const int*   __restrict__ is_init,
                      const float* __restrict__ hx,
                      const float* __restrict__ cx,
                      const __bf16* __restrict__ Wc,
                      const float* __restrict__ biasSum,
                      const float* __restrict__ ln_w,
                      const float* __restrict__ ln_b,
                      float* __restrict__ out,
                      float* __restrict__ hT,
                      float* __restrict__ cT) {
  extern __shared__ char smem[];
  float*  sGates = (float*)(smem + OFF_GATES);   // [16][GSTRIDE]
  __bf16* sAxh   = (__bf16*)(smem + OFF_AXH);    // [16][ASTRIDE]  = [x_t | h]
  float*  sX     = (float*)(smem + OFF_X);       // [16][XSTRIDE]  fp32 x_t
  float*  sC     = (float*)(smem + OFF_C);       // [16][XSTRIDE]  cell state
  float*  sH     = (float*)(smem + OFF_H);       // [16][XSTRIDE]  fp32 h_t
  float*  sBias  = (float*)(smem + OFF_BIAS);    // [GG]

  const int tid  = threadIdx.x;
  const int wave = tid >> 5;
  const int lane = tid & 31;
  const int lm   = lane & 15;      // M (A) / N (B) index within tile
  const int lhi  = lane >> 4;      // K-half select
  const int r    = tid >> 4;       // elementwise row 0..15
  const int cb   = tid & 15;       // elementwise col base
  const int b0   = blockIdx.x * 16;

  // ---- init: bias to LDS, h0/c0 from hx/cx ----
  for (int q = tid; q < GG; q += 256) sBias[q] = biasSum[q];
  for (int j = 0; j < 16; ++j) {
    const int c = cb + j * 16;
    sC[r * XSTRIDE + c] = cx[(size_t)(b0 + r) * HH + c];
    sAxh[r * ASTRIDE + II + c] = (__bf16)hx[(size_t)(b0 + r) * HH + c];
  }
  __syncthreads();

  for (int t = 0; t < TT; ++t) {
    // ---- phase 1: stage x_t (f32 + bf16), apply episode reset to h/c ----
    const int ri = is_init[(size_t)(b0 + r) * TT + t];
    const float* xr = x + ((size_t)(b0 + r) * TT + t) * II;
    if (t + 1 < TT) __builtin_prefetch(xr + II, 0, 0);   // next step's x row
    for (int j = 0; j < 16; ++j) {
      const int c = cb + j * 16;
      const float xv = xr[c];
      sX[r * XSTRIDE + c] = xv;
      sAxh[r * ASTRIDE + c] = (__bf16)xv;
      if (ri) { sAxh[r * ASTRIDE + II + c] = (__bf16)0.0f; sC[r * XSTRIDE + c] = 0.0f; }
    }
    __syncthreads();

    // ---- phase 2: gates[16x1024] = [x|h](16x512) @ Wc^T, bf16 WMMA ----
    v8f acc[8];
    for (int i = 0; i < 8; ++i)
      for (int v = 0; v < 8; ++v) acc[i][v] = 0.0f;

    for (int kb = 0; kb < 16; ++kb) {
      // A frag (16-bit 16x32): lane<16 -> K {0..15}; lane>=16 -> {16..31}
      const __bf16* ap = sAxh + lm * ASTRIDE + kb * 32 + lhi * 16;
      v16bf afrag;
      ((uint4*)&afrag)[0] = *(const uint4*)(ap);
      ((uint4*)&afrag)[1] = *(const uint4*)(ap + 8);
#pragma unroll
      for (int i = 0; i < 8; ++i) {
        // B frag: column N = lane%16 (= Wc row), 16 consecutive K per lane
        const int n = (wave * 8 + i) * 16 + lm;
        const uint4* bp = (const uint4*)(Wc + (size_t)n * KK + kb * 32 + lhi * 16);
        v16bf bfrag;
        ((uint4*)&bfrag)[0] = bp[0];
        ((uint4*)&bfrag)[1] = bp[1];
        acc[i] = __builtin_amdgcn_wmma_f32_16x16x32_bf16(
            false, afrag, false, bfrag, (short)0, acc[i], false, false);
      }
    }
    // D layout: VGPR v -> M = v + 8*lhi, N = lane%16
    for (int i = 0; i < 8; ++i) {
      const int n0 = (wave * 8 + i) * 16 + lm;
#pragma unroll
      for (int v = 0; v < 8; ++v)
        sGates[(lhi * 8 + v) * GSTRIDE + n0] = acc[i][v];
    }
    __syncthreads();

    // ---- phase 3: fused LSTM cell ----
    for (int j = 0; j < 16; ++j) {
      const int col = cb + j * 16;
      const float gi = sGates[r * GSTRIDE + col      ] + sBias[col      ];
      const float gf = sGates[r * GSTRIDE + col + 256] + sBias[col + 256];
      const float gg = sGates[r * GSTRIDE + col + 512] + sBias[col + 512];
      const float go = sGates[r * GSTRIDE + col + 768] + sBias[col + 768];
      const float iv = fsig(gi), fv = fsig(gf), gv = ftanh(gg), ov = fsig(go);
      const float c  = fv * sC[r * XSTRIDE + col] + iv * gv;
      const float h  = ov * ftanh(c);
      sC[r * XSTRIDE + col] = c;
      sH[r * XSTRIDE + col] = h;
      sAxh[r * ASTRIDE + II + col] = (__bf16)h;     // feed next step's A matrix
    }
    __syncthreads();

    // ---- phase 4: LayerNorm + skip-add, one wave per 2 rows ----
    for (int rr = 0; rr < 2; ++rr) {
      const int rw = wave * 2 + rr;
      float vals[8];
      float s = 0.0f;
#pragma unroll
      for (int j = 0; j < 8; ++j) { vals[j] = sH[rw * XSTRIDE + lane + j * 32]; s += vals[j]; }
#pragma unroll
      for (int off = 16; off; off >>= 1) s += __shfl_xor(s, off, 32);
      const float mu = s * (1.0f / 256.0f);
      float vs = 0.0f;
#pragma unroll
      for (int j = 0; j < 8; ++j) { const float d = vals[j] - mu; vs += d * d; }
#pragma unroll
      for (int off = 16; off; off >>= 1) vs += __shfl_xor(vs, off, 32);
      const float inv = rsqrtf(vs * (1.0f / 256.0f) + 1e-5f);
      float* orow = out + ((size_t)(b0 + rw) * TT + t) * HH;
#pragma unroll
      for (int j = 0; j < 8; ++j) {
        const int c = lane + j * 32;
        orow[c] = sX[rw * XSTRIDE + c] + (vals[j] - mu) * inv * ln_w[c] + ln_b[c];
      }
    }
    __syncthreads();
  }

  // ---- final states to workspace for broadcast kernel ----
  for (int j = 0; j < 16; ++j) {
    const int c = cb + j * 16;
    hT[(size_t)(b0 + r) * HH + c] = sH[r * XSTRIDE + c];
    cT[(size_t)(b0 + r) * HH + c] = sC[r * XSTRIDE + c];
  }
}

// ---------------------------------------------------------------------------
// Kernel 3: broadcast hT/cT over T (float4, pure bandwidth)
// ---------------------------------------------------------------------------
__global__ void lstm_bcast_kernel(const float* __restrict__ hT,
                                  const float* __restrict__ cT,
                                  float* __restrict__ outH,
                                  float* __restrict__ outC) {
  const size_t gid = (size_t)blockIdx.x * 256 + threadIdx.x;  // float4 index
  const size_t e = gid * 4;                                    // element index
  const int b = (int)(e >> 16);                                // / (T*H)
  const int j = (int)(e & (HH - 1));                           // % H
  const float4 hv = *(const float4*)(hT + (size_t)b * HH + j);
  const float4 cv = *(const float4*)(cT + (size_t)b * HH + j);
  *(float4*)(outH + e) = hv;
  *(float4*)(outC + e) = cv;
}

// ---------------------------------------------------------------------------
extern "C" void kernel_launch(void* const* d_in, const int* in_sizes, int n_in,
                              void* d_out, int out_size, void* d_ws, size_t ws_size,
                              hipStream_t stream) {
  (void)in_sizes; (void)n_in; (void)out_size; (void)ws_size;
  const float* xp   = (const float*)d_in[0];
  const int*   isi  = (const int*)  d_in[1];
  const float* hx   = (const float*)d_in[2];
  const float* cx   = (const float*)d_in[3];
  const float* Wih  = (const float*)d_in[4];
  const float* Whh  = (const float*)d_in[5];
  const float* bih  = (const float*)d_in[6];
  const float* bhh  = (const float*)d_in[7];
  const float* lnw  = (const float*)d_in[8];
  const float* lnb  = (const float*)d_in[9];

  char* ws = (char*)d_ws;
  __bf16* Wc      = (__bf16*)ws;                                  // 1 MB
  float*  biasSum = (float*)(ws + (1 << 20));                     // 4 KB
  float*  hT      = (float*)(ws + (1 << 20) + 4096);              // 512 KB
  float*  cT      = (float*)(ws + (1 << 20) + 4096 + BN * HH * 4);// 512 KB

  float* out  = (float*)d_out;
  float* outH = out  + (size_t)BN * TT * HH;
  float* outC = outH + (size_t)BN * TT * HH;

  lstm_prep_kernel<<<(GG * KK) / 256, 256, 0, stream>>>(Wih, Whh, bih, bhh, Wc, biasSum);
  lstm_scan_kernel<<<BN / 16, 256, LDS_BYTES, stream>>>(
      xp, isi, hx, cx, Wc, biasSum, lnw, lnb, out, hT, cT);
  lstm_bcast_kernel<<<(int)(((size_t)BN * TT * HH / 4) / 256), 256, 0, stream>>>(
      hT, cT, outH, outC);
}
